// SMILESGenerator_41489384079900
// MI455X (gfx1250) — matrix operand order repaired and checked by reading, not compiled, not run on hardware
//
#include <hip/hip_runtime.h>

// ---------------- problem constants ----------------
#define BB   256     // batch
#define LL   128     // seq len
#define CC   100     // charset
#define HH   512     // lstm dim
#define GG   2048    // 4*H
#define CND  128     // condition dim
#define BT   16      // batch rows per workgroup
#define KP0  128     // padded layer-0 input K (100 -> 128)
#define NOUT 112     // padded output cols (100 -> 112)

// LDS layout (dynamic shared, starts at offset 0):
//   hst : [3][16][512] f16   -> 49152 B
//   cst : [3][16][512] f32   -> 98304 B    (offset 49152)
//   xA  : 2 x [16][128] f16  ->  8192 B    (offset 147456)
#define HST_OFF  0u
#define CST_OFF  (3u * BT * HH * 2u)
#define XA_OFF   (CST_OFF + 3u * BT * HH * 4u)
#define SMEM_SZ  (XA_OFF + 2u * BT * KP0 * 2u)   // 155,648 B

typedef __attribute__((ext_vector_type(16))) _Float16 v16h;
typedef __attribute__((ext_vector_type(8)))  _Float16 v8h;
typedef __attribute__((ext_vector_type(8)))  float    v8f;

// ---------------- WMMA fragment helpers ----------------
// A fragment (16x32 f16, row-major source, row stride = `stride` halves).
// lanes 0-15: M=lane, K={0..7,16..23}; lanes 16-31: M=lane-16, K={8..15,24..31}.
__device__ __forceinline__ v16h load_a_frag(const _Float16* base, int stride,
                                            int kb, int lane) {
  const int r = lane & 15, hi = lane >> 4;
  const _Float16* p = base + r * stride + kb + hi * 8;
  v8h lo = *(const v8h*)p;
  v8h hi8 = *(const v8h*)(p + 16);
  v16h a;
#pragma unroll
  for (int e = 0; e < 8; ++e) { a[e] = lo[e]; a[e + 8] = hi8[e]; }
  return a;
}

// B fragment (32x16 f16). W is [N][Kdim] row-major (gates = x @ W^T):
// lane r = local col n, K={0..7,16..23} (lanes 0-15) / {8..15,24..31} (lanes 16-31).
__device__ __forceinline__ v16h load_b_frag(const _Float16* __restrict__ W,
                                            int Kdim, int nbase, int kb, int lane) {
  const int r = lane & 15, hi = lane >> 4;
  const _Float16* p = W + (size_t)(nbase + r) * Kdim + kb + hi * 8;
  v8h lo = *(const v8h*)p;
  v8h hi8 = *(const v8h*)(p + 16);
  v16h b;
#pragma unroll
  for (int e = 0; e < 8; ++e) { b[e] = lo[e]; b[e + 8] = hi8[e]; }
  return b;
}

__device__ __forceinline__ v8f wmma_acc(v16h a, v16h b, v8f c) {
  return __builtin_amdgcn_wmma_f32_16x16x32_f16(false, a, false, b,
                                                (short)0, c, false, false);
}

__device__ __forceinline__ float sigmoid_(float x) {
  return 1.f / (1.f + __expf(-x));
}
__device__ __forceinline__ float tanh_(float x) {
  x = fminf(fmaxf(x, -15.f), 15.f);
  float e = __expf(2.f * x);
  return (e - 1.f) / (e + 1.f);
}

// Async DMA: one b128 (16B) per lane from global f16 buffer into LDS.
// dsaddr = wave LDS base + VGPR value; tracked by ASYNCcnt.
__device__ __forceinline__ void stage_x_async(const _Float16* __restrict__ Xf16,
                                              int t, int b0, int tid,
                                              unsigned buf_off) {
  const int m = tid >> 4;            // local batch row (16B x 16 thr = one row)
  const int k = (tid & 15) * 8;      // half index within row
  const _Float16* gp = Xf16 + ((size_t)t * BB + b0 + m) * KP0 + k;
  const unsigned lds = buf_off + (unsigned)tid * 16u;
  asm volatile("global_load_async_to_lds_b128 %0, %1, off"
               :: "v"(lds), "v"((unsigned long long)(uintptr_t)gp)
               : "memory");
}
__device__ __forceinline__ void wait_async0() {
  asm volatile("s_wait_asynccnt 0x0" ::: "memory");
}

// ---------------- prep kernels ----------------
__global__ void cvt_pad_kernel(const float* __restrict__ src, _Float16* __restrict__ dst,
                               int Nsrc, int rows, int Ksrc, int Kdst) {
  int i = blockIdx.x * blockDim.x + threadIdx.x;
  int total = rows * Kdst;
  if (i >= total) return;
  int n = i / Kdst, k = i % Kdst;
  float v = (n < Nsrc && k < Ksrc) ? src[(size_t)n * Ksrc + k] : 0.f;
  dst[i] = (_Float16)v;
}

__global__ void bias_add_kernel(const float* __restrict__ a, const float* __restrict__ b,
                                float* __restrict__ o, int n) {
  int i = blockIdx.x * blockDim.x + threadIdx.x;
  if (i < n) o[i] = a[i] + b[i];
}

// Pre-shifted f16 inputs: Xf16[t][b][k] = din[b][t==0?0:t-1][k<C? k : pad0]
__global__ void xcvt_kernel(const float* __restrict__ din, _Float16* __restrict__ xf) {
  int i = blockIdx.x * blockDim.x + threadIdx.x;
  const int total = LL * BB * KP0;
  if (i >= total) return;
  int t = i / (BB * KP0);
  int rem = i % (BB * KP0);
  int b = rem / KP0, k = rem % KP0;
  int ts = (t == 0) ? 0 : t - 1;
  float v = (k < CC) ? din[((size_t)b * LL + ts) * CC + k] : 0.f;
  xf[i] = (_Float16)v;
}

// h0/c0 = relu(cond @ W^T + b) per layer; h as f16, c as f32.
__global__ void init_state_kernel(const float* __restrict__ cond,
                                  const float* __restrict__ hw, const float* __restrict__ hb,
                                  const float* __restrict__ cw, const float* __restrict__ cb,
                                  _Float16* __restrict__ h0, float* __restrict__ c0) {
  int i = blockIdx.x * blockDim.x + threadIdx.x;
  if (i >= 3 * BB * HH) return;
  int l = i / (BB * HH);
  int rem = i % (BB * HH);
  int b = rem / HH, h = rem % HH;
  const float* wh = hw + ((size_t)l * HH + h) * CND;
  const float* wc = cw + ((size_t)l * HH + h) * CND;
  const float* cd = cond + (size_t)b * CND;
  float sh = 0.f, sc = 0.f;
#pragma unroll 4
  for (int d = 0; d < CND; ++d) {
    float x = cd[d];
    sh += x * wh[d];
    sc += x * wc[d];
  }
  sh += hb[l * HH + h];
  sc += cb[l * HH + h];
  h0[i] = (_Float16)fmaxf(sh, 0.f);
  c0[i] = fmaxf(sc, 0.f);
}

// ---------------- persistent decoder ----------------
// 16 blocks x 256 threads (8 wave32). Block owns batch rows [blk*16,+16).
// Full time loop inside; h/c state LDS-resident; x tiles double-buffered
// via global_load_async_to_lds (ASYNCcnt) one step ahead.
__global__ __launch_bounds__(256) void smiles_decoder_kernel(
    const _Float16* __restrict__ Xf16,    // [L][B][128] pre-shifted f16
    const _Float16* __restrict__ Wih0,    // [2048,128] padded
    const _Float16* __restrict__ Whh0,    // [2048,512]
    const _Float16* __restrict__ Wih1,
    const _Float16* __restrict__ Whh1,
    const _Float16* __restrict__ Wih2,
    const _Float16* __restrict__ Whh2,
    const _Float16* __restrict__ WoutH,   // [112,512] padded
    const float* __restrict__ biasg,      // [3][2048] (bih+bhh)
    const float* __restrict__ bout,       // [100]
    const _Float16* __restrict__ h0g,     // [3][B][512]
    const float* __restrict__ c0g,        // [3][B][512]
    float* __restrict__ out)              // [B,L,C] fp32
{
  extern __shared__ char smem[];
  _Float16* hst = (_Float16*)(smem + HST_OFF);   // [3][16][512] f16
  float*    cst = (float*)(smem + CST_OFF);      // [3][16][512] f32

  const int tid  = threadIdx.x;
  const int wave = tid >> 5;
  const int lane = tid & 31;
  const int r    = lane & 15;
  const int hi   = lane >> 4;
  const int b0   = blockIdx.x * BT;

  // ---- load initial state into LDS ----
  for (int i = tid; i < 3 * BT * HH; i += 256) {
    int l = i / (BT * HH);
    int rem = i % (BT * HH);
    int m = rem / HH, h = rem % HH;
    size_t gidx = ((size_t)l * BB + b0 + m) * HH + h;
    hst[i] = h0g[gidx];
    cst[i] = c0g[gidx];
  }
  // kick off async DMA of x tile for step 0 into buffer 0
  stage_x_async(Xf16, 0, b0, tid, XA_OFF);
  __syncthreads();

  for (int t = 0; t < LL; ++t) {
    wait_async0();        // x tile for step t landed in LDS
    __syncthreads();
    if (t + 1 < LL)       // overlap next step's DMA with this step's compute
      stage_x_async(Xf16, t + 1, b0, tid, XA_OFF + ((unsigned)(t + 1) & 1u) * (BT * KP0 * 2u));

    const _Float16* xA = (const _Float16*)(smem + XA_OFF + (unsigned)(t & 1) * (BT * KP0 * 2u));

#pragma unroll
    for (int l = 0; l < 3; ++l) {
      const _Float16* Wi  = (l == 0) ? Wih0 : ((l == 1) ? Wih1 : Wih2);
      const _Float16* Wh  = (l == 0) ? Whh0 : ((l == 1) ? Whh1 : Whh2);
      const int       kin = (l == 0) ? KP0 : HH;
      const _Float16* Ain = (l == 0) ? xA : (hst + (l - 1) * BT * HH);
      const _Float16* Ah  = hst + l * BT * HH;
      const float*    bl  = biasg + l * GG;

      // each wave: 4 h-ranges; per range one 16-col tile of each gate i,f,g,o
#pragma unroll 1
      for (int k4 = 0; k4 < 4; ++k4) {
        const int hcb = (wave + 8 * k4) * 16;  // h-column base, 0..496
        v8f ai = {}, af = {}, ag = {}, ao = {};

        // input contribution: A = x (or h of layer below), K = kin
#pragma unroll 2
        for (int kb = 0; kb < kin; kb += 32) {
          __builtin_prefetch(Wi + (size_t)(hcb + r) * kin + kb + 64, 0, 1);
          v16h a  = load_a_frag(Ain, kin, kb, lane);
          v16h bi = load_b_frag(Wi, kin, 0 * HH + hcb, kb, lane);
          v16h bf = load_b_frag(Wi, kin, 1 * HH + hcb, kb, lane);
          v16h bg = load_b_frag(Wi, kin, 2 * HH + hcb, kb, lane);
          v16h bo = load_b_frag(Wi, kin, 3 * HH + hcb, kb, lane);
          ai = wmma_acc(a, bi, ai);
          af = wmma_acc(a, bf, af);
          ag = wmma_acc(a, bg, ag);
          ao = wmma_acc(a, bo, ao);
        }
        // recurrent contribution: A = h_l (old), K = 512
#pragma unroll 2
        for (int kb = 0; kb < HH; kb += 32) {
          __builtin_prefetch(Wh + (size_t)(hcb + r) * HH + kb + 64, 0, 1);
          v16h a  = load_a_frag(Ah, HH, kb, lane);
          v16h bi = load_b_frag(Wh, HH, 0 * HH + hcb, kb, lane);
          v16h bf = load_b_frag(Wh, HH, 1 * HH + hcb, kb, lane);
          v16h bg = load_b_frag(Wh, HH, 2 * HH + hcb, kb, lane);
          v16h bo = load_b_frag(Wh, HH, 3 * HH + hcb, kb, lane);
          ai = wmma_acc(a, bi, ai);
          af = wmma_acc(a, bf, af);
          ag = wmma_acc(a, bg, ag);
          ao = wmma_acc(a, bo, ao);
        }

        // ---- LSTM elementwise on the WMMA D layout ----
        const int hc = hcb + r;  // this lane's h column
        const float b_i = bl[0 * HH + hc];
        const float b_f = bl[1 * HH + hc];
        const float b_g = bl[2 * HH + hc];
        const float b_o = bl[3 * HH + hc];
#pragma unroll
        for (int j = 0; j < 8; ++j) {
          const int m = j + 8 * hi;  // local batch row
          const int sidx = (l * BT + m) * HH + hc;
          float iv = ai[j] + b_i;
          float fv = af[j] + b_f;
          float gv = ag[j] + b_g;
          float ov = ao[j] + b_o;
          float cold = cst[sidx];
          float cn = sigmoid_(fv) * cold + sigmoid_(iv) * tanh_(gv);
          float hn = sigmoid_(ov) * tanh_(cn);
          cst[sidx] = cn;
          hst[sidx] = (_Float16)hn;
        }
      }
      __syncthreads();  // full h_l visible before layer l+1 consumes it
    }

    // ---- output projection: logits = h2 @ Wout^T + bout (7 tiles) ----
    if (wave < 7) {
      const _Float16* Ah = hst + 2 * BT * HH;
      v8f acc = {};
#pragma unroll 2
      for (int kb = 0; kb < HH; kb += 32) {
        v16h a = load_a_frag(Ah, HH, kb, lane);
        v16h b = load_b_frag(WoutH, HH, wave * 16, kb, lane);
        acc = wmma_acc(a, b, acc);
      }
      const int cc = wave * 16 + r;
      if (cc < CC) {
        const float bo_ = bout[cc];
#pragma unroll
        for (int j = 0; j < 8; ++j) {
          const int m = j + 8 * hi;
          out[((size_t)(b0 + m) * LL + t) * CC + cc] = acc[j] + bo_;
        }
      }
    }
    __syncthreads();  // protect hst reuse across steps
  }
}

// ---------------- host launch ----------------
extern "C" void kernel_launch(void* const* d_in, const int* in_sizes, int n_in,
                              void* d_out, int out_size, void* d_ws, size_t ws_size,
                              hipStream_t stream) {
  (void)in_sizes; (void)n_in; (void)out_size; (void)ws_size;
  const float* cond = (const float*)d_in[0];
  const float* din  = (const float*)d_in[1];
  const float* hw   = (const float*)d_in[2];
  const float* hb   = (const float*)d_in[3];
  const float* cw   = (const float*)d_in[4];
  const float* cb   = (const float*)d_in[5];
  const float* wih0 = (const float*)d_in[6];
  const float* whh0 = (const float*)d_in[7];
  const float* bih0 = (const float*)d_in[8];
  const float* bhh0 = (const float*)d_in[9];
  const float* wih1 = (const float*)d_in[10];
  const float* whh1 = (const float*)d_in[11];
  const float* bih1 = (const float*)d_in[12];
  const float* bhh1 = (const float*)d_in[13];
  const float* wih2 = (const float*)d_in[14];
  const float* whh2 = (const float*)d_in[15];
  const float* bih2 = (const float*)d_in[16];
  const float* bhh2 = (const float*)d_in[17];
  const float* wout = (const float*)d_in[18];
  const float* bout = (const float*)d_in[19];
  float* out = (float*)d_out;

  char* ws = (char*)d_ws;
  size_t off = 0;
  _Float16* Wih0h = (_Float16*)(ws + off); off += (size_t)GG * KP0 * 2;
  _Float16* Whh0h = (_Float16*)(ws + off); off += (size_t)GG * HH * 2;
  _Float16* Wih1h = (_Float16*)(ws + off); off += (size_t)GG * HH * 2;
  _Float16* Whh1h = (_Float16*)(ws + off); off += (size_t)GG * HH * 2;
  _Float16* Wih2h = (_Float16*)(ws + off); off += (size_t)GG * HH * 2;
  _Float16* Whh2h = (_Float16*)(ws + off); off += (size_t)GG * HH * 2;
  _Float16* WoutH = (_Float16*)(ws + off); off += (size_t)NOUT * HH * 2;
  float*    biasg = (float*)(ws + off);    off += (size_t)3 * GG * 4;
  _Float16* h0g   = (_Float16*)(ws + off); off += (size_t)3 * BB * HH * 2;
  float*    c0g   = (float*)(ws + off);    off += (size_t)3 * BB * HH * 4;
  _Float16* Xf16  = (_Float16*)(ws + off); off += (size_t)LL * BB * KP0 * 2;

  const int TPB = 256;
  auto blocks = [](size_t n) { return (unsigned)((n + 255) / 256); };

  // weight conversion / padding
  cvt_pad_kernel<<<blocks((size_t)GG * KP0), TPB, 0, stream>>>(wih0, Wih0h, GG, GG, CC, KP0);
  cvt_pad_kernel<<<blocks((size_t)GG * HH), TPB, 0, stream>>>(whh0, Whh0h, GG, GG, HH, HH);
  cvt_pad_kernel<<<blocks((size_t)GG * HH), TPB, 0, stream>>>(wih1, Wih1h, GG, GG, HH, HH);
  cvt_pad_kernel<<<blocks((size_t)GG * HH), TPB, 0, stream>>>(whh1, Whh1h, GG, GG, HH, HH);
  cvt_pad_kernel<<<blocks((size_t)GG * HH), TPB, 0, stream>>>(wih2, Wih2h, GG, GG, HH, HH);
  cvt_pad_kernel<<<blocks((size_t)GG * HH), TPB, 0, stream>>>(whh2, Whh2h, GG, GG, HH, HH);
  cvt_pad_kernel<<<blocks((size_t)NOUT * HH), TPB, 0, stream>>>(wout, WoutH, CC, NOUT, HH, HH);

  // fused biases
  bias_add_kernel<<<blocks(GG), TPB, 0, stream>>>(bih0, bhh0, biasg + 0 * GG, GG);
  bias_add_kernel<<<blocks(GG), TPB, 0, stream>>>(bih1, bhh1, biasg + 1 * GG, GG);
  bias_add_kernel<<<blocks(GG), TPB, 0, stream>>>(bih2, bhh2, biasg + 2 * GG, GG);

  // pre-shifted f16 inputs for async DMA
  xcvt_kernel<<<blocks((size_t)LL * BB * KP0), TPB, 0, stream>>>(din, Xf16);

  // initial states
  init_state_kernel<<<blocks((size_t)3 * BB * HH), TPB, 0, stream>>>(
      cond, hw, hb, cw, cb, h0g, c0g);

  // persistent decoder: 16 blocks, LDS = state + double-buffered x tiles
  smiles_decoder_kernel<<<dim3(BB / BT), dim3(TPB), SMEM_SZ, stream>>>(
      Xf16, Wih0h, Whh0h, Wih1h, Whh1h, Wih2h, Whh2h, WoutH,
      biasg, bout, h0g, c0g, out);
}
